// MoE_48095043780864
// MI455X (gfx1250) — matrix operand order repaired and checked by reading, not compiled, not run on hardware
//
#include <hip/hip_runtime.h>
#include <hip/hip_bf16.h>

// ---------------------------------------------------------------------------
// MoE (2x ResNet-20 experts + soft-topk gate) forward for gfx1250 (MI455X).
// Convolutions = implicit GEMM on v_wmma_f32_16x16x32_bf16, fp32 accumulate.
// One wave owns a 16(M=pixels) x (NT*16)(N=channels) tile; B fragments are
// pre-swizzled for single vector loads; epilogue (BN/bias+residual+ReLU) is
// fused and uses b128 vector loads/stores (M-tiles are 16 consecutive pixels).
// Only the 6 (TIn,KS,NT) combos the network actually executes are
// instantiated: (f32,3,1) stem/gate, (bf16,3,1/2/4) 3x3 stages,
// (bf16,1,2/4) projection shortcuts.
// ---------------------------------------------------------------------------

typedef __attribute__((ext_vector_type(16))) __bf16 v16bf;
typedef __attribute__((ext_vector_type(8)))  __bf16 v8bf;
typedef __attribute__((ext_vector_type(8)))  float  v8f;

__device__ __forceinline__ __bf16 f2bf(float f) {
  unsigned u = __builtin_bit_cast(unsigned, f);
  u += 0x7FFFu + ((u >> 16) & 1u);               // round-to-nearest-even
  unsigned short h = (unsigned short)(u >> 16);
  return __builtin_bit_cast(__bf16, h);
}
__device__ __forceinline__ float bf2f(__bf16 b) {
  unsigned short h = __builtin_bit_cast(unsigned short, b);
  unsigned u = ((unsigned)h) << 16;
  return __builtin_bit_cast(float, u);
}
__device__ __forceinline__ float ld_as_f32(const float*  p) { return *p; }
__device__ __forceinline__ float ld_as_f32(const __bf16* p) { return bf2f(*p); }

// ---------------------------------------------------------------------------
// Weight prep: W[Cout][K] fp32 -> swizzled bf16 B fragments laid out so each
// lane's 16-element WMMA B fragment is contiguous (32B, one vector load):
//   Bsw[((kb*Ntiles + nt)*32 + lane)*16 + j]  with
//   k = kb*32 + (lane>>4)*16 + j,  co = nt*16 + (lane&15)
// ---------------------------------------------------------------------------
__global__ __launch_bounds__(256)
void prep_weights_kernel(const float* __restrict__ W, __bf16* __restrict__ Bsw,
                         int K, int Ksteps, int Ntiles, int Cout) {
  int i = blockIdx.x * blockDim.x + threadIdx.x;
  if (i >= Ksteps * Ntiles * 512) return;
  int j    = i & 15;
  int lane = (i >> 4) & 31;
  int rest = i >> 9;
  int nt   = rest % Ntiles;
  int kb   = rest / Ntiles;
  int k    = (kb << 5) + (lane >> 4) * 16 + j;
  int co   = nt * 16 + (lane & 15);
  float v  = (k < K) ? W[co * K + k] : 0.f;
  Bsw[i] = f2bf(v);
}

// ---------------------------------------------------------------------------
// Implicit-GEMM conv. SAME padding (pad_lo passed; high pad via bounds check).
// Epilogue: y = acc*gamma + beta (gamma==null -> 1), +resid, optional ReLU,
// stored bf16 with one b128 vector store per lane per N-tile.
// ---------------------------------------------------------------------------
template <typename TIn, int KS, int NT>
__global__ __launch_bounds__(128)
void conv_wmma_kernel(const TIn* __restrict__ in, const __bf16* __restrict__ Bsw,
                      __bf16* out, const __bf16* resid,
                      const float* __restrict__ gamma, const float* __restrict__ beta,
                      int NI, int Cin, int Hin, int Win,
                      int Cout, int Hout, int Wout,
                      int strideV, int pad, int relu) {
  const int lane = threadIdx.x & 31;
  const int wave = threadIdx.x >> 5;
  const int HWo  = Hout * Wout;
  const int M    = NI * HWo;
  const int mtile = blockIdx.x * 4 + wave;
  if (mtile * 16 >= M) return;
  const int hl = lane >> 4;      // lane half (K split per ISA A/B layouts)
  const int mr = lane & 15;

  // A-fragment row for this lane (tile row M = mr); hoisted im2col bases.
  const int am   = mtile * 16 + mr;
  const int aimg = am / HWo;
  const int arem = am - aimg * HWo;
  const int aoy  = arem / Wout;
  const int aox  = arem - aoy * Wout;
  const int iy0  = aoy * strideV - pad;
  const int ix0  = aox * strideV - pad;
  const TIn* inB = in + (size_t)aimg * Cin * Hin * Win;

  const int Ktot   = Cin * KS * KS;
  const int Kfull  = Ktot >> 5;            // unchecked K-steps
  const int Ksteps = (Ktot + 31) >> 5;
  const int Ntiles = gridDim.y * NT;       // Cout / 16
  const __bf16* bptr = Bsw + ((size_t)(blockIdx.y * NT) * 32 + lane) * 16;
  const size_t  bstep = (size_t)Ntiles * 512;   // Bsw elems per K-step

  v8f acc[NT];
  const v8f vzero = {0.f, 0.f, 0.f, 0.f, 0.f, 0.f, 0.f, 0.f};
#pragma unroll
  for (int t = 0; t < NT; ++t) acc[t] = vzero;

  auto kstep = [&](int kb, bool chk) {
    const int k0 = kb << 5;
    v16bf a;
    // A: 16x32 bf16; lane hl=0 -> K {0..15}, hl=1 -> K {16..31}
#pragma unroll
    for (int j = 0; j < 16; ++j) {
      const int k = k0 + hl * 16 + j;
      float v = 0.f;
      if (!chk || k < Ktot) {
        int ci, iy, ix;
        if (KS == 1) {            // 1x1: k==ci, never out of spatial range
          ci = k; iy = iy0; ix = ix0;
          v = ld_as_f32(inB + (ci * Hin + iy) * Win + ix);
        } else {
          ci = k / (KS * KS);
          int r  = k - ci * (KS * KS);
          int ky = r / KS;
          iy = iy0 + ky;
          ix = ix0 + (r - ky * KS);
          if ((unsigned)iy < (unsigned)Hin && (unsigned)ix < (unsigned)Win)
            v = ld_as_f32(inB + (ci * Hin + iy) * Win + ix);
        }
      }
      a[j] = f2bf(v);
    }
    // B: one 32B vector load per N-tile; NT WMMAs share the A fragment.
#pragma unroll
    for (int t = 0; t < NT; ++t) {
      v16bf b = *(const v16bf*)(bptr + (size_t)kb * bstep + (size_t)t * 512);
      acc[t] = __builtin_amdgcn_wmma_f32_16x16x32_bf16(
          false, a, false, b, (short)0, acc[t], false, false);
    }
  };

  for (int kb = 0; kb < Kfull; ++kb) kstep(kb, false);
  if (Kfull < Ksteps) kstep(Kfull, true);

  // Epilogue. C/D layout: VGPR r -> tile row hl*8+r, column = lane%16.
  // A 16-pixel M-tile never crosses an image (HWo % 16 == 0), so each lane's
  // 8 rows are 8 consecutive NCHW pixels -> one b128 load/store.
  const int gm0 = mtile * 16 + hl * 8;
  const int img = gm0 / HWo;
  const int p0  = gm0 - img * HWo;
#pragma unroll
  for (int t = 0; t < NT; ++t) {
    const int co = (blockIdx.y * NT + t) * 16 + mr;
    const float g  = gamma ? gamma[co] : 1.f;
    const float bb = beta  ? beta[co]  : 0.f;
    const size_t idx0 = (size_t)(img * Cout + co) * HWo + p0;
    v8bf stv;
    if (resid) {
      v8bf rv = *(const v8bf*)(resid + idx0);   // in-place safe: same lane RMW
#pragma unroll
      for (int r = 0; r < 8; ++r) {
        float v = acc[t][r] * g + bb + bf2f(rv[r]);
        if (relu) v = v > 0.f ? v : 0.f;
        stv[r] = f2bf(v);
      }
    } else {
#pragma unroll
      for (int r = 0; r < 8; ++r) {
        float v = acc[t][r] * g + bb;
        if (relu) v = v > 0.f ? v : 0.f;
        stv[r] = f2bf(v);
      }
    }
    *(v8bf*)(out + idx0) = stv;
  }
}

// ---------------------------------------------------------------------------
// Global average pool over HW: in bf16 [rows][HW] -> out fp32 [rows].
// ---------------------------------------------------------------------------
__global__ __launch_bounds__(256)
void pool_kernel(const __bf16* __restrict__ in, float* __restrict__ out, int HW) {
  __shared__ float sm[256];
  const int row = blockIdx.x;
  const __bf16* p = in + (long)row * HW;
  float s = 0.f;
  for (int i = threadIdx.x; i < HW; i += blockDim.x) s += bf2f(p[i]);
  sm[threadIdx.x] = s;
  __syncthreads();
  for (int off = blockDim.x >> 1; off > 0; off >>= 1) {
    if ((int)threadIdx.x < off) sm[threadIdx.x] += sm[threadIdx.x + off];
    __syncthreads();
  }
  if (threadIdx.x == 0) out[row] = sm[0] / (float)HW;
}

// Small dense layer: out[n][c] = feat[n]·w[c] + b[c]
__global__ __launch_bounds__(256)
void fc_kernel(const float* __restrict__ feat, const float* __restrict__ w,
               const float* __restrict__ b, float* __restrict__ out,
               int N, int Cin, int Cout) {
  int i = blockIdx.x * blockDim.x + threadIdx.x;
  if (i >= N * Cout) return;
  int n = i / Cout, c = i - n * Cout;
  float s = b ? b[c] : 0.f;
  const float* f  = feat + n * Cin;
  const float* wr = w + c * Cin;
  for (int k = 0; k < Cin; ++k) s += f[k] * wr[k];
  out[i] = s;
}

__global__ void zero_kernel(float* p, int n) {
  int i = blockIdx.x * blockDim.x + threadIdx.x;
  if (i < n) p[i] = 0.f;
}

// softmax + soft-topk (n=2, K=2, tau1=tau2=1e-5) + importance accumulation
__global__ __launch_bounds__(256)
void gate_kernel(const float* __restrict__ logits, float* __restrict__ gates,
                 float* importance, int N) {
  int n = blockIdx.x * blockDim.x + threadIdx.x;
  if (n >= N) return;
  float l0 = logits[2 * n], l1 = logits[2 * n + 1];
  float mx = fmaxf(l0, l1);
  float e0 = __expf(l0 - mx), e1 = __expf(l1 - mx);
  float inv = 1.f / (e0 + e1);
  float s0 = e0 * inv, s1 = e1 * inv;
  const float it = 1e5f;  // 1/tau
  auto sig = [](float x) { return 1.f / (1.f + __expf(-x)); };
  float r0 = 1.f + (sig(0.f) + sig(-(s0 - s1) * it) - 0.5f);
  float r1 = 1.f + (sig(-(s1 - s0) * it) + sig(0.f) - 0.5f);
  float a0 = sig((2.5f - r0) * it);
  float a1 = sig((2.5f - r1) * it);
  float g0 = a0 * s0, g1 = a1 * s1;
  gates[2 * n]     = g0;
  gates[2 * n + 1] = g1;
  atomicAdd(&importance[0], g0);
  atomicAdd(&importance[1], g1);
}

// cv_squared (ddof=1 over 2 values) * 0.01
__global__ void loss_kernel(const float* importance, float* out_loss) {
  if (threadIdx.x == 0 && blockIdx.x == 0) {
    float a = importance[0], b = importance[1];
    float m = 0.5f * (a + b);
    float var = (a - m) * (a - m) + (b - m) * (b - m);  // /(2-1)
    out_loss[0] = 0.01f * var / (m * m + 1e-10f);
  }
}

// y = g0*e1 + g1*e2
__global__ __launch_bounds__(256)
void combine_kernel(const float* __restrict__ e1, const float* __restrict__ e2,
                    const float* __restrict__ gates, float* __restrict__ y,
                    int N, int C) {
  int i = blockIdx.x * blockDim.x + threadIdx.x;
  if (i >= N * C) return;
  int n = i / C;
  y[i] = gates[2 * n] * e1[i] + gates[2 * n + 1] * e2[i];
}

// ---------------------------------------------------------------------------
// Host-side orchestration
// ---------------------------------------------------------------------------
static inline int ceil_div(int a, int b) { return (a + b - 1) / b; }

static void run_conv(hipStream_t stream, const void* in, bool in_f32,
                     const float* W, __bf16* wbuf, __bf16* out, __bf16* resid,
                     const float* gamma, const float* beta,
                     int NI, int Cin, int Hin, int Cout, int strideV, int ks,
                     int relu) {
  const int Hout   = (Hin + strideV - 1) / strideV;      // SAME
  const int K      = Cin * ks * ks;
  const int Ksteps = (K + 31) >> 5;
  const int Ntiles = Cout / 16;
  const int pad    = (ks == 3 && strideV == 1) ? 1 : 0;  // XLA SAME low-pad
  prep_weights_kernel<<<ceil_div(Ksteps * Ntiles * 512, 256), 256, 0, stream>>>(
      W, wbuf, K, Ksteps, Ntiles, Cout);
  const int NT = (Cout >= 64) ? 4 : (Cout >= 32) ? 2 : 1;
  const int M  = NI * Hout * Hout;
  dim3 grid(ceil_div(M, 64), Cout / (16 * NT));
  dim3 block(128);

#define CWK(T_, KS_, NT_)                                                    \
  conv_wmma_kernel<T_, KS_, NT_><<<grid, block, 0, stream>>>(                \
      (const T_*)in, wbuf, out, resid, gamma, beta, NI, Cin, Hin, Hin,       \
      Cout, Hout, Hout, strideV, pad, relu)

  // Only the combos the network executes are instantiated.
  if (in_f32) {
    CWK(float, 3, 1);                       // stem + gate conv (3->16, 3x3)
  } else if (ks == 3) {
    if (NT == 1)      CWK(__bf16, 3, 1);    // stage-1 3x3 (Cout=16)
    else if (NT == 2) CWK(__bf16, 3, 2);    // stage-2 3x3 (Cout=32)
    else              CWK(__bf16, 3, 4);    // stage-3 3x3 (Cout=64)
  } else {
    if (NT == 2)      CWK(__bf16, 1, 2);    // projection 16->32
    else              CWK(__bf16, 1, 4);    // projection 32->64
  }
#undef CWK
}

struct BlockP {
  const float *w1, *g1, *b1, *w2, *g2, *b2, *wp, *gp, *bp;
  int s; bool proj;
};
struct ExpertP {
  const float *w0, *g0, *b0, *fcw, *fcb;
  BlockP blk[9];
};

// Leaf order assumption: top-level dict in insertion order; nested pytrees in
// jax tree-flatten order (dict keys sorted; tuple (params, stride) in order,
// stride is a 1-element int leaf which we skip).
static int extract_expert(void* const* d_in, int i, ExpertP& E) {
  static const int projb[9]   = {0, 0, 0, 1, 0, 0, 1, 0, 0};
  static const int strides[9] = {1, 1, 1, 2, 1, 1, 2, 1, 1};
  E.b0 = (const float*)d_in[i++];
  for (int b = 0; b < 9; ++b) {
    BlockP& B = E.blk[b];
    B.proj = projb[b] != 0;
    B.s = strides[b];
    if (B.proj) {  // sorted: b1,b2,bp,g1,g2,gp,w1,w2,wp
      B.b1 = (const float*)d_in[i++]; B.b2 = (const float*)d_in[i++];
      B.bp = (const float*)d_in[i++]; B.g1 = (const float*)d_in[i++];
      B.g2 = (const float*)d_in[i++]; B.gp = (const float*)d_in[i++];
      B.w1 = (const float*)d_in[i++]; B.w2 = (const float*)d_in[i++];
      B.wp = (const float*)d_in[i++];
    } else {       // sorted: b1,b2,g1,g2,w1,w2
      B.b1 = (const float*)d_in[i++]; B.b2 = (const float*)d_in[i++];
      B.g1 = (const float*)d_in[i++]; B.g2 = (const float*)d_in[i++];
      B.w1 = (const float*)d_in[i++]; B.w2 = (const float*)d_in[i++];
      B.wp = B.gp = B.bp = nullptr;
    }
    i++;  // stride int leaf
  }
  E.fcb = (const float*)d_in[i++];
  E.fcw = (const float*)d_in[i++];
  E.g0  = (const float*)d_in[i++];
  E.w0  = (const float*)d_in[i++];
  return i;
}

static void run_expert(hipStream_t s, const float* x, const ExpertP& E,
                       __bf16* A0, __bf16* A1, __bf16* A2, __bf16* wbuf,
                       float* pool, float* eout) {
  __bf16 *h = A0, *t1 = A1, *t2 = A2;
  // stem: conv3x3 3->16, bn, relu
  run_conv(s, x, true, E.w0, wbuf, h, nullptr, E.g0, E.b0, 512, 3, 32, 16, 1, 3, 1);
  int Cin = 16, Hin = 32;
  for (int b = 0; b < 9; ++b) {
    const BlockP& B = E.blk[b];
    const int Cout = (b < 3) ? 16 : (b < 6 ? 32 : 64);
    const int Hout = Hin / B.s;
    // conv1: h -> t1 (bn1, relu)
    run_conv(s, h, false, B.w1, wbuf, t1, nullptr, B.g1, B.b1,
             512, Cin, Hin, Cout, B.s, 3, 1);
    __bf16* r = h;
    if (B.proj) {  // projection shortcut: h -> t2 (bnp)
      run_conv(s, h, false, B.wp, wbuf, t2, nullptr, B.gp, B.bp,
               512, Cin, Hin, Cout, B.s, 1, 0);
      r = t2;
    }
    // conv2: t1 -> r, bn2 + residual(r) + relu, in-place residual
    run_conv(s, t1, false, B.w2, wbuf, r, r, B.g2, B.b2,
             512, Cout, Hout, Cout, 1, 3, 1);
    if (B.proj) { __bf16* tmp = h; h = t2; t2 = tmp; }
    Cin = Cout; Hin = Hout;
  }
  // global average pool [512,64,8,8] -> [512,64], then FC 64->10
  pool_kernel<<<512 * 64, 128, 0, s>>>(h, pool, 64);
  fc_kernel<<<ceil_div(512 * 10, 256), 256, 0, s>>>(pool, E.fcw, E.fcb, eout,
                                                    512, 64, 10);
}

extern "C" void kernel_launch(void* const* d_in, const int* in_sizes, int n_in,
                              void* d_out, int out_size, void* d_ws,
                              size_t ws_size, hipStream_t stream) {
  (void)in_sizes; (void)n_in; (void)out_size; (void)ws_size;

  // --- input leaves ---
  const float* x   = (const float*)d_in[0];
  // d_in[1] = train (unused on eval path)
  const float* gcb = (const float*)d_in[2];  // gate cb [16]
  const float* gcw = (const float*)d_in[3];  // gate cw [16,3,3,3]
  const float* glb = (const float*)d_in[4];  // gate lb [2]
  const float* glw = (const float*)d_in[5];  // gate lw [2,16]
  ExpertP E1, E2;
  int idx = extract_expert(d_in, 6, E1);
  extract_expert(d_in, idx, E2);

  // --- workspace carve-up ---
  size_t off = 0;
  auto carve = [&](size_t bytes) -> char* {
    char* p = (char*)d_ws + off;
    off = (off + bytes + 255) & ~(size_t)255;
    return p;
  };
  const size_t ACT = (size_t)512 * 16 * 32 * 32 * sizeof(__bf16);  // 16.8 MB
  __bf16* A0   = (__bf16*)carve(ACT);
  __bf16* A1   = (__bf16*)carve(ACT);
  __bf16* A2   = (__bf16*)carve(ACT);
  __bf16* wbuf = (__bf16*)carve((size_t)576 * 64 * sizeof(__bf16));
  float* gpool = (float*)carve(512 * 16 * sizeof(float));
  float* glog  = (float*)carve(512 * 2 * sizeof(float));
  float* gates = (float*)carve(512 * 2 * sizeof(float));
  float* imp   = (float*)carve(2 * sizeof(float));
  float* epool = (float*)carve(512 * 64 * sizeof(float));
  float* e1out = (float*)carve(512 * 10 * sizeof(float));
  float* e2out = (float*)carve(512 * 10 * sizeof(float));

  // --- gate network ---
  // conv3x3 3->16 + bias (gamma=null -> 1.0, beta=cb), relu -> A0
  run_conv(stream, x, true, gcw, wbuf, A0, nullptr, nullptr, gcb,
           512, 3, 32, 16, 1, 3, 1);
  pool_kernel<<<512 * 16, 256, 0, stream>>>(A0, gpool, 1024);
  fc_kernel<<<ceil_div(512 * 2, 256), 256, 0, stream>>>(gpool, glw, glb, glog,
                                                        512, 16, 2);
  zero_kernel<<<1, 32, 0, stream>>>(imp, 2);
  gate_kernel<<<ceil_div(512, 256), 256, 0, stream>>>(glog, gates, imp, 512);

  // --- experts ---
  run_expert(stream, x, E1, A0, A1, A2, wbuf, epool, e1out);
  run_expert(stream, x, E2, A0, A1, A2, wbuf, epool, e2out);

  // --- combine + loss ---
  float* yout = (float*)d_out;
  combine_kernel<<<ceil_div(512 * 10, 256), 256, 0, stream>>>(e1out, e2out,
                                                              gates, yout,
                                                              512, 10);
  loss_kernel<<<1, 1, 0, stream>>>(imp, yout + 512 * 10);
}
